// DLRMModel_47356309405934
// MI455X (gfx1250) — matrix operand order repaired and checked by reading, not compiled, not run on hardware
//
#include <hip/hip_runtime.h>
#include <hip/hip_bf16.h>

typedef __bf16 bf16_t;
typedef __attribute__((ext_vector_type(8)))  __bf16 v8bf;
typedef __attribute__((ext_vector_type(16))) __bf16 v16bf;
typedef __attribute__((ext_vector_type(8)))  float  v8f;

typedef int v4i_t __attribute__((vector_size(16)));
typedef __attribute__((address_space(1))) v4i_t g_v4i;   // global
typedef __attribute__((address_space(3))) v4i_t l_v4i;   // LDS

#define BATCH   16384
#define EMB_DIM 128
#define N_TBL   26
#define VOCAB   100000
#define N_CONT  13
#define MLP_IN  3456   // 128 * 27

// ---------- bf16 helpers (RNE, via bit ops) ----------
static __device__ __forceinline__ bf16_t f2bf(float f) {
    unsigned u = __builtin_bit_cast(unsigned, f);
    unsigned r = u + 0x7FFFu + ((u >> 16) & 1u);
    unsigned short h = (unsigned short)(r >> 16);
    return __builtin_bit_cast(bf16_t, h);
}
static __device__ __forceinline__ float bf2f(bf16_t b) {
    unsigned short h = __builtin_bit_cast(unsigned short, b);
    unsigned u = ((unsigned)h) << 16;
    return __builtin_bit_cast(float, u);
}

// ---------- CDNA5 async global->LDS copy (ASYNCcnt-tracked, no VGPR round trip) ----------
static __device__ __forceinline__ void async_copy_b128(const bf16_t* g, bf16_t* l) {
#if __has_builtin(__builtin_amdgcn_global_load_async_to_lds_b128)
    // signature: (v4i addrspace(1)* src, v4i addrspace(3)* dst, imm offset, imm cpol)
    __builtin_amdgcn_global_load_async_to_lds_b128(
        (g_v4i*)g, (l_v4i*)l, 0, 0);
#else
    unsigned loff = (unsigned)(__UINTPTR_TYPE__)(l_v4i*)l;
    asm volatile("global_load_async_to_lds_b128 %0, %1, off"
                 :: "v"(loff), "v"(g) : "memory");
#endif
}

template <int N>
static __device__ __forceinline__ void wait_asynccnt_le() {
#if __has_builtin(__builtin_amdgcn_s_wait_asynccnt)
    __builtin_amdgcn_s_wait_asynccnt(N);
#else
    asm volatile("s_wait_asynccnt %0" :: "i"(N) : "memory");
#endif
}

// ---------- weight convert + transpose: W[K,N] f32 -> Wt[N,K] bf16 ----------
__global__ void wconv_kernel(const float* __restrict__ W, bf16_t* __restrict__ Wt,
                             int K, int N) {
    int i = blockIdx.x * blockDim.x + threadIdx.x;
    if (i >= K * N) return;
    int k = i / N, n = i % N;
    Wt[(size_t)n * K + k] = f2bf(W[i]);
}

// ---------- fused bottom MLP (13->128) + 26-table gather -> h0 [B, 3456] bf16 ----------
__global__ void bottom_gather_kernel(const float* __restrict__ cont,
                                     const int*   __restrict__ cat,
                                     const float* __restrict__ emb,
                                     const float* __restrict__ Wc,
                                     const float* __restrict__ bc,
                                     bf16_t* __restrict__ h0) {
    const int b = blockIdx.x;     // one batch row per block
    const int t = threadIdx.x;    // 0..127
    float s = bc[t];
#pragma unroll
    for (int i = 0; i < N_CONT; ++i)
        s += cont[b * N_CONT + i] * Wc[i * EMB_DIM + t];
    bf16_t* row = h0 + (size_t)b * MLP_IN;
    row[t] = f2bf(s);
#pragma unroll
    for (int T = 0; T < N_TBL; ++T) {
        int idx = cat[b * N_TBL + T];
        const float* src = emb + ((size_t)T * VOCAB + (size_t)idx) * EMB_DIM;
        row[EMB_DIM + T * EMB_DIM + t] = f2bf(src[t]);
    }
}

// ---------- bf16 WMMA GEMM: C[M,N] = act(A[M,K] @ Bt[N,K]^T + bias) ----------
// block = 256 threads = 8 waves; block tile 128(M) x 256(N); wave tile 64x64.
// Double-buffered LDS filled by async global->LDS loads (6 per thread per tile).
__global__ __launch_bounds__(256)
void gemm_bf16_kernel(const bf16_t* __restrict__ A, const bf16_t* __restrict__ Bt,
                      const float* __restrict__ bias, bf16_t* __restrict__ C,
                      int M, int N, int K, int doRelu) {
    __shared__ bf16_t ldsA[2][128 * 32];   // 2 x  8 KB
    __shared__ bf16_t ldsB[2][256 * 32];   // 2 x 16 KB

    const int tid   = threadIdx.x;
    const int lane  = tid & 31;
    const int wave  = tid >> 5;
    const int waveM = wave & 1;         // 0..1  (64-row halves)
    const int waveN = wave >> 1;        // 0..3  (64-col quarters)
    const int M0 = blockIdx.x * 128;
    const int N0 = blockIdx.y * 256;
    const int m0 = M0 + waveM * 64;
    const int n0 = N0 + waveN * 64;

    v8f acc[4][4] = {};

    // staging assignments (whole block cooperates)
    const int aRow  = tid >> 1, aHalf = tid & 1;          // 128 rows x 2 halves of 16 bf16
    const bf16_t* aSrc = A  + (size_t)(M0 + aRow) * K + aHalf * 16;
    const bf16_t* bSrc = Bt + (size_t)(N0 + tid) * K;     // 256 rows x 32 bf16
    const int aOff = aRow * 32 + aHalf * 16;
    const int bOff = tid * 32;

    // fragment lane geometry (ISA 16-bit A / B layouts, wave32)
    const int l15 = lane & 15;
    const int aK  = (lane >> 4) * 8;    // A: lanes 0-15 -> K 0..7/16..23 ; 16-31 -> 8..15/24..31
    const int bK  = (lane >> 4) * 8;    // B: lanes 0-15 -> K 0..7/16..23 ; 16-31 -> 8..15/24..31

    const int nsteps = K >> 5;

    // prologue: async-stage tile 0 into buffer 0
    {
        const bf16_t* as = aSrc;
        const bf16_t* bs = bSrc;
        async_copy_b128(as,      &ldsA[0][aOff]);
        async_copy_b128(as + 8,  &ldsA[0][aOff + 8]);
        async_copy_b128(bs,      &ldsB[0][bOff]);
        async_copy_b128(bs + 8,  &ldsB[0][bOff + 8]);
        async_copy_b128(bs + 16, &ldsB[0][bOff + 16]);
        async_copy_b128(bs + 24, &ldsB[0][bOff + 24]);
    }

    for (int s = 0; s < nsteps; ++s) {
        const int buf = s & 1;
        if (s + 1 < nsteps) {
            // issue next tile's async loads into the other buffer, then wait until
            // only those 6 remain outstanding (current tile complete, in-order).
            const int k1 = (s + 1) << 5;
            const bf16_t* as = aSrc + k1;
            const bf16_t* bs = bSrc + k1;
            async_copy_b128(as,      &ldsA[buf ^ 1][aOff]);
            async_copy_b128(as + 8,  &ldsA[buf ^ 1][aOff + 8]);
            async_copy_b128(bs,      &ldsB[buf ^ 1][bOff]);
            async_copy_b128(bs + 8,  &ldsB[buf ^ 1][bOff + 8]);
            async_copy_b128(bs + 16, &ldsB[buf ^ 1][bOff + 16]);
            async_copy_b128(bs + 24, &ldsB[buf ^ 1][bOff + 24]);
            wait_asynccnt_le<6>();
        } else {
            wait_asynccnt_le<0>();
        }
        __syncthreads();   // all waves' tile-s data now resident in ldsA/ldsB[buf]

        // ---- build fragments from LDS ----
        v16bf af[4], bfr[4];
#pragma unroll
        for (int mi = 0; mi < 4; ++mi) {
            int row = waveM * 64 + mi * 16 + l15;
            v8bf lo = *(const v8bf*)&ldsA[buf][row * 32 + aK];
            v8bf hi = *(const v8bf*)&ldsA[buf][row * 32 + aK + 16];
            af[mi] = __builtin_shufflevector(lo, hi,
                       0,1,2,3,4,5,6,7,8,9,10,11,12,13,14,15);
        }
#pragma unroll
        for (int ni = 0; ni < 4; ++ni) {
            int col = waveN * 64 + ni * 16 + l15;
            v8bf blo = *(const v8bf*)&ldsB[buf][col * 32 + bK];
            v8bf bhi = *(const v8bf*)&ldsB[buf][col * 32 + bK + 16];
            bfr[ni] = __builtin_shufflevector(blo, bhi,
                       0,1,2,3,4,5,6,7,8,9,10,11,12,13,14,15);
        }
        // ---- 16 WMMAs per wave per K-step ----
#pragma unroll
        for (int mi = 0; mi < 4; ++mi)
#pragma unroll
            for (int ni = 0; ni < 4; ++ni)
                acc[mi][ni] = __builtin_amdgcn_wmma_f32_16x16x32_bf16(
                    false, af[mi], false, bfr[ni],
                    (short)0, acc[mi][ni], false, false);
        __syncthreads();   // everyone done reading buf before it is re-staged
    }

    // ---- epilogue: bias + (optional) relu, store bf16 ----
#pragma unroll
    for (int mi = 0; mi < 4; ++mi) {
#pragma unroll
        for (int ni = 0; ni < 4; ++ni) {
            int n = n0 + ni * 16 + l15;
            int mb = m0 + mi * 16 + ((lane >> 4) << 3);
            float bv = bias[n];
#pragma unroll
            for (int r = 0; r < 8; ++r) {
                float v = acc[mi][ni][r] + bv;
                if (doRelu) v = v > 0.f ? v : 0.f;
                C[(size_t)(mb + r) * N + n] = f2bf(v);
            }
        }
    }
}

// ---------- head: out[b] = sigmoid(h3[b,:] . Wout + b_out), one wave per row ----------
__global__ void head_kernel(const bf16_t* __restrict__ h3, const float* __restrict__ Wout,
                            const float* __restrict__ bout, float* __restrict__ out) {
    const int wave = threadIdx.x >> 5, lane = threadIdx.x & 31;
    const int b = blockIdx.x * 8 + wave;
    const bf16_t* row = h3 + (size_t)b * 256;
    float s = 0.f;
#pragma unroll
    for (int j = 0; j < 8; ++j) {
        int k = lane + j * 32;
        s += bf2f(row[k]) * Wout[k];
    }
#pragma unroll
    for (int o = 16; o > 0; o >>= 1) s += __shfl_xor(s, o, 32);
    if (lane == 0) out[b] = 1.f / (1.f + __expf(-(s + bout[0])));
}

extern "C" void kernel_launch(void* const* d_in, const int* in_sizes, int n_in,
                              void* d_out, int out_size, void* d_ws, size_t ws_size,
                              hipStream_t stream) {
    (void)in_sizes; (void)n_in; (void)out_size; (void)ws_size;
    const float* cont = (const float*)d_in[0];
    const int*   cat  = (const int*)  d_in[1];
    const float* emb  = (const float*)d_in[2];
    const float* Wc   = (const float*)d_in[3];
    const float* bc   = (const float*)d_in[4];
    const float* W1   = (const float*)d_in[5];
    const float* b1   = (const float*)d_in[6];
    const float* W2   = (const float*)d_in[7];
    const float* b2   = (const float*)d_in[8];
    const float* W3   = (const float*)d_in[9];
    const float* b3   = (const float*)d_in[10];
    const float* Wout = (const float*)d_in[11];
    const float* bout = (const float*)d_in[12];
    float* out = (float*)d_out;

    char* ws = (char*)d_ws;
    size_t off = 0;
    auto take = [&](size_t bytes) -> char* {
        char* p = ws + off;
        off += (bytes + 255) & ~(size_t)255;
        return p;
    };
    bf16_t* W1t = (bf16_t*)take((size_t)MLP_IN * 1024 * 2);
    bf16_t* W2t = (bf16_t*)take((size_t)1024 * 512 * 2);
    bf16_t* W3t = (bf16_t*)take((size_t)512 * 256 * 2);
    bf16_t* h0  = (bf16_t*)take((size_t)BATCH * MLP_IN * 2);
    bf16_t* h1  = (bf16_t*)take((size_t)BATCH * 1024 * 2);
    bf16_t* h2  = (bf16_t*)take((size_t)BATCH * 512 * 2);
    bf16_t* h3  = (bf16_t*)take((size_t)BATCH * 256 * 2);

    // 1) weight convert+transpose to bf16 [N,K]
    wconv_kernel<<<(MLP_IN * 1024 + 255) / 256, 256, 0, stream>>>(W1, W1t, MLP_IN, 1024);
    wconv_kernel<<<(1024 * 512 + 255) / 256, 256, 0, stream>>>(W2, W2t, 1024, 512);
    wconv_kernel<<<(512 * 256 + 255) / 256, 256, 0, stream>>>(W3, W3t, 512, 256);

    // 2) bottom MLP + embedding gather -> h0
    bottom_gather_kernel<<<BATCH, EMB_DIM, 0, stream>>>(cont, cat, emb, Wc, bc, h0);

    // 3) top MLP via bf16 WMMA
    dim3 blk(256);
    dim3 g1(BATCH / 128, 1024 / 256);
    gemm_bf16_kernel<<<g1, blk, 0, stream>>>(h0, W1t, b1, h1, BATCH, 1024, MLP_IN, 1);
    dim3 g2(BATCH / 128, 512 / 256);
    gemm_bf16_kernel<<<g2, blk, 0, stream>>>(h1, W2t, b2, h2, BATCH, 512, 1024, 1);
    dim3 g3(BATCH / 128, 256 / 256);
    gemm_bf16_kernel<<<g3, blk, 0, stream>>>(h2, W3t, b3, h3, BATCH, 256, 512, 1);

    // 4) head GEMV + sigmoid
    head_kernel<<<BATCH / 8, 256, 0, stream>>>(h3, Wout, bout, out);
}
